// QuantumEEGNet_23278722744731
// MI455X (gfx1250) — compile-verified, matrix operand order checked by reading, not run on hardware
//
#include <hip/hip_runtime.h>
#include <hip/hip_bf16.h>

// ---------------------------------------------------------------------------
// QuantumEEGNet forward for MI455X (gfx1250, wave32).
//  - conv2/conv3 (the two K=1024 GEMMs, ~47 of ~50 GFLOP) run on the matrix
//    pipe via V_WMMA_F32_16X16X32_F16 (f16 A/B, f32 accumulate).
//  - Each GEMM wave computes a 64Mx32N tile (4 M-subtiles x 2 N-tiles = 8
//    accumulators): B fragments loaded once per K-step and reused 4x ->
//    ~21 flops/byte from L2 (h1p=30MB f16 + 64KB weights fit in 192MB L2).
//  - im2col folded into per-lane fragment address math (no LDS staging).
//  - 9-qubit statevector: 1 sample/wave, 512 amps = 16 regs/lane,
//    cross-lane pairs via __shfl_xor (wave32).
// ---------------------------------------------------------------------------

typedef __attribute__((ext_vector_type(16))) _Float16 v16h;
typedef __attribute__((ext_vector_type(8)))  float    v8f;
typedef _Float16 f16;

#define DEV __device__ __forceinline__

DEV float eluf(float x) { return x > 0.f ? x : (__expf(x) - 1.f); }

// ======================= conv1: (1024,1,4,216) x (16,1,1,64) -> (1024,16,4,153)
__global__ void conv1_kernel(const float* __restrict__ x,
                             const float* __restrict__ w1,
                             float* __restrict__ h1) {
  int t = blockIdx.x * 256 + threadIdx.x;
  const int TOT = 1024 * 16 * 4 * 153;
  if (t >= TOT) return;
  int w  = t % 153;
  int h  = (t / 153) & 3;
  int oc = (t / 612) & 15;
  int b  = t / 9792;
  const float* xr = x + (b * 4 + h) * 216 + w;
  const float* wr = w1 + oc * 64;
  float acc = 0.f;
#pragma unroll 8
  for (int k = 0; k < 64; ++k) acc += xr[k] * wr[k];
  h1[t] = acc;
}

// ======================= batch-norm statistics (sum / sumsq per channel)
__global__ void zero_stats_kernel(float* __restrict__ s) {
  s[threadIdx.x] = 0.f;   // 256 floats
}

__global__ void bn_stats_kernel(const float* __restrict__ src, int C, int HW,
                                long long total_per_c, int splits,
                                float* __restrict__ sumO, float* __restrict__ sqO) {
  __shared__ float ssum[256], ssq[256];
  int c  = blockIdx.x % C;
  int sp = blockIdx.x / C;
  long long chunk = (total_per_c + splits - 1) / splits;
  long long beg = (long long)sp * chunk;
  long long end = beg + chunk; if (end > total_per_c) end = total_per_c;
  float s = 0.f, q = 0.f;
  for (long long t = beg + threadIdx.x; t < end; t += 256) {
    long long n = t / HW, hw = t % HW;
    float v = src[(n * C + c) * HW + hw];
    s += v; q += v * v;
  }
  ssum[threadIdx.x] = s; ssq[threadIdx.x] = q;
  __syncthreads();
  for (int o = 128; o > 0; o >>= 1) {
    if ((int)threadIdx.x < o) {
      ssum[threadIdx.x] += ssum[threadIdx.x + o];
      ssq[threadIdx.x]  += ssq[threadIdx.x + o];
    }
    __syncthreads();
  }
  if (threadIdx.x == 0) { atomicAdd(&sumO[c], ssum[0]); atomicAdd(&sqO[c], ssq[0]); }
}

// ======================= weights fp32 -> f16 (conv2 & conv3, 32768 each)
__global__ void cvt_w_kernel(const float* __restrict__ w2, const float* __restrict__ w3,
                             f16* __restrict__ w2h, f16* __restrict__ w3h) {
  int t = blockIdx.x * 256 + threadIdx.x;
  if (t < 32768) { w2h[t] = (f16)w2[t]; w3h[t] = (f16)w3[t]; }
}

// ======================= BN1 + ELU + pad((0,1),(16,17)) -> f16 (1024,16,5,186)
__global__ void bn1_pad_kernel(const float* __restrict__ h1,
                               const float* __restrict__ stats,
                               f16* __restrict__ h1p) {
  int t = blockIdx.x * 256 + threadIdx.x;
  const int TOT = 1024 * 16 * 5 * 186;
  if (t >= TOT) return;
  int col = t % 186;
  int row = (t / 186) % 5;
  int c   = (t / 930) & 15;
  int b   = t / 14880;
  float v = 0.f;
  if (row < 4 && col >= 16 && col < 169) {
    const float cnt = 1024.f * 4.f * 153.f;
    float mu  = stats[c] / cnt;
    float var = stats[16 + c] / cnt - mu * mu;
    float xv  = h1[((b * 16 + c) * 4 + row) * 153 + (col - 16)];
    v = eluf((xv - mu) * rsqrtf(var + 1e-5f));
  }
  h1p[t] = (f16)v;
}

// ======================= conv2 as WMMA GEMM: M=634880, N=32, K=1024
// A(im2col): k = c*64 + kh*32 + kw over h1p(1024,16,5,186); B: w2h[n*1024+k].
// 16-bit A-frag layout (ISA 7.12.2): M=lane&15, K-half by lane>>4,
// pair p holds K = 2p + (p>=4?8:0) + 8*(lane>=16). C/D: elem r <-> M=r+8*(lane>=16), N=lane&15.
// Each wave: 4 M-subtiles x 2 N-tiles = 8 accumulators; B frags reused 4x per K-step.
__global__ void conv2_wmma_kernel(const f16* __restrict__ A,
                                  const f16* __restrict__ Bw,
                                  float* __restrict__ h2) {
  int wave = blockIdx.x * 8 + ((int)threadIdx.x >> 5);
  int lane = threadIdx.x & 31;
  const int WT = 9920;                 // 634880 / 64, exact
  if (wave >= WT) return;              // wave-uniform: EXEC stays all-ones for WMMA
  int ml = lane & 15;
  int hi = lane >> 4;
  int mrow0 = wave * 64;
  v8f acc[8];
#pragma unroll
  for (int i = 0; i < 8; ++i) acc[i] = (v8f){};
  // per-subtile A row base (f16 elements), excluding the per-K-step (c,kh) term:
  // full offset = ((b*16+c)*5 + oh+kh)*186 + ow = (b*80+oh)*186 + ow  +  (c*5+kh)*186
  int abase[4];
#pragma unroll
  for (int s = 0; s < 4; ++s) {
    int m  = mrow0 + s * 16 + ml;
    int ow = m % 155;
    int oh = (m / 155) & 3;
    int b  = m / 620;
    abase[s] = (b * 80 + oh) * 186 + ow;
  }
  for (int ks = 0; ks < 32; ++ks) {          // K = 32 * 32
    int c = ks >> 1, kh = ks & 1;
    int koff = (c * 5 + kh) * 186;
    const f16* brow0 = Bw + ml * 1024 + ks * 32;
    const f16* brow1 = Bw + (16 + ml) * 1024 + ks * 32;
    v16h bf0, bf1;
#pragma unroll
    for (int p = 0; p < 8; ++p) {
      int k = 2 * p + (p >= 4 ? 8 : 0) + hi * 8;   // consecutive K pair (k, k+1)
      bf0[2*p] = brow0[k];  bf0[2*p+1] = brow0[k + 1];
      bf1[2*p] = brow1[k];  bf1[2*p+1] = brow1[k + 1];
    }
#pragma unroll
    for (int s = 0; s < 4; ++s) {
      const f16* arow = A + abase[s] + koff;
      v16h af;
#pragma unroll
      for (int p = 0; p < 8; ++p) {
        int k = 2 * p + (p >= 4 ? 8 : 0) + hi * 8;
        af[2*p] = arow[k];  af[2*p+1] = arow[k + 1];
      }
      acc[2*s]   = __builtin_amdgcn_wmma_f32_16x16x32_f16(false, af, false, bf0, (short)0, acc[2*s],   false, false);
      acc[2*s+1] = __builtin_amdgcn_wmma_f32_16x16x32_f16(false, af, false, bf1, (short)0, acc[2*s+1], false, false);
    }
  }
#pragma unroll
  for (int s = 0; s < 4; ++s) {
#pragma unroll
    for (int r = 0; r < 8; ++r) {
      int mm  = mrow0 + s * 16 + r + hi * 8;
      int ow2 = mm % 155;
      int oh2 = (mm / 155) & 3;
      int b2  = mm / 620;
      h2[((b2 * 32 + ml)        * 4 + oh2) * 155 + ow2] = acc[2*s][r];
      h2[((b2 * 32 + (16 + ml)) * 4 + oh2) * 155 + ow2] = acc[2*s+1][r];
    }
  }
}

// ======================= BN2 + ELU + maxpool(2,4) + pad((4,3),(2,1)) -> f16 (1024,32,9,41)
__global__ void bn2_pool_pad_kernel(const float* __restrict__ h2,
                                    const float* __restrict__ stats,
                                    f16* __restrict__ p2p) {
  int t = blockIdx.x * 256 + threadIdx.x;
  const int TOT = 1024 * 32 * 9 * 41;
  if (t >= TOT) return;
  int col = t % 41;
  int row = (t / 41) % 9;
  int c   = (t / 369) & 31;
  int b   = t / 11808;
  float v = 0.f;
  if (row >= 4 && row < 6 && col >= 2 && col < 40) {
    int ph = row - 4, pw = col - 2;
    const float cnt = 1024.f * 4.f * 155.f;
    float mu  = stats[64 + c] / cnt;
    float var = stats[96 + c] / cnt - mu * mu;
    float inv = rsqrtf(var + 1e-5f);
    const float* base = h2 + ((b * 32 + c) * 4 + 2 * ph) * 155 + 4 * pw;
    float mx = -1e30f;
#pragma unroll
    for (int r = 0; r < 2; ++r)
#pragma unroll
      for (int q = 0; q < 4; ++q)
        mx = fmaxf(mx, eluf((base[r * 155 + q] - mu) * inv));
    v = mx;
  }
  p2p[t] = (f16)v;
}

// ======================= conv3 as WMMA GEMM: M=77824, N=32, K=1024
// A(im2col): k = c*32 + kh*4 + kw over p2p(1024,32,9,41); B: w3h[n*1024+k].
// Same 64Mx32N per-wave tiling as conv2.
__global__ void conv3_wmma_kernel(const f16* __restrict__ A,
                                  const f16* __restrict__ Bw,
                                  float* __restrict__ h3) {
  int wave = blockIdx.x * 8 + ((int)threadIdx.x >> 5);
  int lane = threadIdx.x & 31;
  const int WT = 1216;                 // 77824 / 64, exact
  if (wave >= WT) return;
  int ml = lane & 15;
  int hi = lane >> 4;
  int mrow0 = wave * 64;
  v8f acc[8];
#pragma unroll
  for (int i = 0; i < 8; ++i) acc[i] = (v8f){};
  // full offset = ((b*32+c)*9 + oh+kh)*41 + ow+kw = (b*288+oh)*41+ow + c*369 + kh*41+kw
  int abase[4];
#pragma unroll
  for (int s = 0; s < 4; ++s) {
    int m  = mrow0 + s * 16 + ml;
    int ow = m % 38;
    int oh = (m / 38) & 1;
    int b  = m / 76;
    abase[s] = (b * 288 + oh) * 41 + ow;
  }
  for (int ks = 0; ks < 32; ++ks) {          // one input channel per K-step
    int koff = ks * 369;
    const f16* brow0 = Bw + ml * 1024 + ks * 32;
    const f16* brow1 = Bw + (16 + ml) * 1024 + ks * 32;
    v16h bf0, bf1;
#pragma unroll
    for (int p = 0; p < 8; ++p) {
      int k = 2 * p + (p >= 4 ? 8 : 0) + hi * 8;
      bf0[2*p] = brow0[k];  bf0[2*p+1] = brow0[k + 1];
      bf1[2*p] = brow1[k];  bf1[2*p+1] = brow1[k + 1];
    }
#pragma unroll
    for (int s = 0; s < 4; ++s) {
      const f16* arow = A + abase[s] + koff;
      v16h af;
#pragma unroll
      for (int p = 0; p < 8; ++p) {
        int k  = 2 * p + (p >= 4 ? 8 : 0) + hi * 8;
        int kh = k >> 2, kw = k & 3;           // pair (k,k+1) stays in one row
        const f16* ap = arow + kh * 41 + kw;
        af[2*p] = ap[0];  af[2*p+1] = ap[1];
      }
      acc[2*s]   = __builtin_amdgcn_wmma_f32_16x16x32_f16(false, af, false, bf0, (short)0, acc[2*s],   false, false);
      acc[2*s+1] = __builtin_amdgcn_wmma_f32_16x16x32_f16(false, af, false, bf1, (short)0, acc[2*s+1], false, false);
    }
  }
#pragma unroll
  for (int s = 0; s < 4; ++s) {
#pragma unroll
    for (int r = 0; r < 8; ++r) {
      int mm  = mrow0 + s * 16 + r + hi * 8;
      int ow2 = mm % 38;
      int oh2 = (mm / 38) & 1;
      int b2  = mm / 76;
      h3[((b2 * 32 + ml)        * 2 + oh2) * 38 + ow2] = acc[2*s][r];
      h3[((b2 * 32 + (16 + ml)) * 2 + oh2) * 38 + ow2] = acc[2*s+1][r];
    }
  }
}

// ======================= BN3 + ELU + maxpool(2,4): (1024,32,2,38) -> angles (1024,32,9)
__global__ void bn3_pool_kernel(const float* __restrict__ h3,
                                const float* __restrict__ stats,
                                float* __restrict__ ang) {
  int t = blockIdx.x * 256 + threadIdx.x;
  const int TOT = 1024 * 32 * 9;
  if (t >= TOT) return;
  int j = t % 9;
  int c = (t / 9) & 31;
  int b = t / 288;
  const float cnt = 1024.f * 2.f * 38.f;
  float mu  = stats[128 + c] / cnt;
  float var = stats[160 + c] / cnt - mu * mu;
  float inv = rsqrtf(var + 1e-5f);
  const float* base = h3 + ((b * 32 + c) * 2) * 38 + 4 * j;
  float mx = -1e30f;
#pragma unroll
  for (int r = 0; r < 2; ++r)
#pragma unroll
    for (int q = 0; q < 4; ++q)
      mx = fmaxf(mx, eluf((base[r * 38 + q] - mu) * inv));
  ang[t] = mx;
}

// ======================= 9-qubit statevector, 1 sample per wave32
// amplitude index a = lane*16 + j  (lane -> bits 8..4, j -> bits 3..0)
DEV void apply_ry(float st[16], int lane, int wire, float th) {
  float s, c;
  __sincosf(0.5f * th, &s, &c);
  int p = 8 - wire;                 // paired bit position
  if (p >= 4) {                     // cross-lane pair
    int lm  = 1 << (p - 4);
    int bit = (lane >> (p - 4)) & 1;
#pragma unroll
    for (int j = 0; j < 16; ++j) {
      float pv = __shfl_xor(st[j], lm, 32);
      st[j] = bit ? (s * pv + c * st[j]) : (c * st[j] - s * pv);
    }
  } else {                          // in-register pair
    int msk = 1 << p;
#pragma unroll
    for (int j = 0; j < 16; ++j) {
      if (j & msk) continue;
      float a0 = st[j], b0 = st[j | msk];
      st[j]       = c * a0 - s * b0;
      st[j | msk] = s * a0 + c * b0;
    }
  }
}

DEV void apply_cnot(float st[16], int lane, int pc, int pt) {
  if (pt >= 4) {
    int lm = 1 << (pt - 4);
#pragma unroll
    for (int j = 0; j < 16; ++j) {
      float pv = __shfl_xor(st[j], lm, 32);   // all lanes participate
      int cbit = (pc >= 4) ? ((lane >> (pc - 4)) & 1) : ((j >> pc) & 1);
      st[j] = cbit ? pv : st[j];
    }
  } else {
    int tm = 1 << pt;
#pragma unroll
    for (int j = 0; j < 16; ++j) {
      if (j & tm) continue;
      int cbit = (pc >= 4) ? ((lane >> (pc - 4)) & 1) : ((j >> pc) & 1);
      if (cbit) { float tmp = st[j]; st[j] = st[j | tm]; st[j | tm] = tmp; }
    }
  }
}

__global__ void quantum_kernel(const float* __restrict__ ang,
                               const float* __restrict__ qw,
                               float* __restrict__ qz) {
  int wave = blockIdx.x * 8 + ((int)threadIdx.x >> 5);   // 4096 blocks * 8 = 32768 samples
  int lane = threadIdx.x & 31;
  float st[16];
#pragma unroll
  for (int j = 0; j < 16; ++j) st[j] = 0.f;
  if (lane == 0) st[0] = 1.f;
  const float* a = ang + wave * 9;
#pragma unroll
  for (int i = 0; i < 9; ++i) apply_ry(st, lane, i, a[i]);
  for (int L = 0; L < 4; ++L) {
#pragma unroll
    for (int i = 0; i < 9; ++i) apply_cnot(st, lane, 8 - i, 8 - ((i + 1) % 9));
#pragma unroll
    for (int i = 0; i < 9; ++i) apply_ry(st, lane, i, qw[L * 9 + i]);
  }
  float z[9];
#pragma unroll
  for (int i = 0; i < 9; ++i) z[i] = 0.f;
#pragma unroll
  for (int j = 0; j < 16; ++j) {
    float p = st[j] * st[j];
#pragma unroll
    for (int i = 0; i < 9; ++i) {
      int pi  = 8 - i;
      int bit = (pi >= 4) ? ((lane >> (pi - 4)) & 1) : ((j >> pi) & 1);
      z[i] += bit ? -p : p;
    }
  }
#pragma unroll
  for (int i = 0; i < 9; ++i)
    for (int off = 16; off; off >>= 1) z[i] += __shfl_xor(z[i], off, 32);
  if (lane == 0) {
#pragma unroll
    for (int i = 0; i < 9; ++i) qz[wave * 9 + i] = z[i];
  }
}

// ======================= FC: (1024,288) @ (288,5)^T + b -> (1024,5)
__global__ void fc_kernel(const float* __restrict__ qz, const float* __restrict__ fw,
                          const float* __restrict__ fb, float* __restrict__ out) {
  int t = blockIdx.x * 256 + threadIdx.x;
  if (t >= 5120) return;
  int o = t % 5;
  int b = t / 5;
  const float* qr = qz + b * 288;
  const float* wr = fw + o * 288;
  float acc = fb[o];
#pragma unroll 8
  for (int k = 0; k < 288; ++k) acc += qr[k] * wr[k];
  out[t] = acc;
}

// ===========================================================================
extern "C" void kernel_launch(void* const* d_in, const int* in_sizes, int n_in,
                              void* d_out, int out_size, void* d_ws, size_t ws_size,
                              hipStream_t stream) {
  const float* x   = (const float*)d_in[0];
  const float* w1  = (const float*)d_in[1];
  const float* w2  = (const float*)d_in[2];
  const float* w3  = (const float*)d_in[3];
  const float* qw  = (const float*)d_in[4];
  const float* fcw = (const float*)d_in[5];
  const float* fcb = (const float*)d_in[6];
  float* out = (float*)d_out;

  char* ws = (char*)d_ws;
  size_t off = 0;
  auto alloc = [&](size_t bytes) -> char* {
    char* p = ws + off;
    off += (bytes + 255) & ~size_t(255);
    return p;
  };
  float* stats = (float*)alloc(256 * 4);                          // [0..15]=sum1 [16..31]=sq1 [64..95]=sum2 [96..127]=sq2 [128..159]=sum3 [160..191]=sq3
  f16*   w2h   = (f16*)alloc(32768 * 2);
  f16*   w3h   = (f16*)alloc(32768 * 2);
  f16*   h1p   = (f16*)alloc((size_t)1024 * 16 * 5 * 186 * 2);    // 30.5 MB
  f16*   p2p   = (f16*)alloc((size_t)1024 * 32 * 9 * 41 * 2);     // 24.2 MB
  float* ang   = (float*)alloc((size_t)1024 * 32 * 9 * 4);        // 1.2 MB
  float* qz    = (float*)alloc((size_t)1024 * 288 * 4);           // 1.2 MB
  char*  big   = alloc((size_t)1024 * 32 * 4 * 155 * 4);          // 81.3 MB, reused 3x
  float* h1raw = (float*)big;   // (1024,16,4,153)  dead after bn1_pad
  float* h2raw = (float*)big;   // (1024,32,4,155)  dead after bn2_pool_pad
  float* h3raw = (float*)big;   // (1024,32,2,38)   dead after bn3_pool
  (void)ws_size; (void)in_sizes; (void)n_in; (void)out_size;

  zero_stats_kernel<<<1, 256, 0, stream>>>(stats);
  conv1_kernel<<<(1024 * 16 * 4 * 153 + 255) / 256, 256, 0, stream>>>(x, w1, h1raw);
  bn_stats_kernel<<<16 * 64, 256, 0, stream>>>(h1raw, 16, 4 * 153, (long long)1024 * 4 * 153, 64,
                                               stats + 0, stats + 16);
  bn1_pad_kernel<<<(1024 * 16 * 5 * 186 + 255) / 256, 256, 0, stream>>>(h1raw, stats, h1p);
  cvt_w_kernel<<<128, 256, 0, stream>>>(w2, w3, w2h, w3h);

  conv2_wmma_kernel<<<1240, 256, 0, stream>>>(h1p, w2h, h2raw);          // 9920 waves, 64Mx32N each
  bn_stats_kernel<<<32 * 64, 256, 0, stream>>>(h2raw, 32, 4 * 155, (long long)1024 * 4 * 155, 64,
                                               stats + 64, stats + 96);
  bn2_pool_pad_kernel<<<(1024 * 32 * 9 * 41 + 255) / 256, 256, 0, stream>>>(h2raw, stats, p2p);

  conv3_wmma_kernel<<<152, 256, 0, stream>>>(p2p, w3h, h3raw);           // 1216 waves
  bn_stats_kernel<<<32 * 16, 256, 0, stream>>>(h3raw, 32, 2 * 38, (long long)1024 * 2 * 38, 16,
                                               stats + 128, stats + 160);
  bn3_pool_kernel<<<(1024 * 32 * 9 + 255) / 256, 256, 0, stream>>>(h3raw, stats, ang);

  quantum_kernel<<<4096, 256, 0, stream>>>(ang, qw, qz);
  fc_kernel<<<20, 256, 0, stream>>>(qz, fcw, fcb, out);
}
